// TimeSeriesMamba_83915071029837
// MI455X (gfx1250) — matrix operand order repaired and checked by reading, not compiled, not run on hardware
//
#include <hip/hip_runtime.h>
#include <cstdint>
#include <cstddef>

// ---------------- model dimensions (fixed by the reference) ----------------
#define BATCH   4
#define SEQ     512
#define DIN     4
#define DMODEL  512
#define NLAYER  4
#define DSTATE  64
#define DCONV   4
#define DINNER  1024      // EXP * DMODEL
#define DTRANK  32        // (DMODEL+15)/16
#define HIDDEN  1408      // glu hidden
#define MROWS   (BATCH*SEQ)   // 2048

#define CDIV(a,b) (((a)+(b)-1)/(b))

typedef __bf16 bf16;
typedef __attribute__((ext_vector_type(16))) __bf16 v16bf;
typedef __attribute__((ext_vector_type(8)))  __bf16 v8bf;
typedef __attribute__((ext_vector_type(8)))  float  v8f;
typedef __attribute__((ext_vector_type(4)))  int    v4i;

// ---------------------------------------------------------------------------
// CDNA5 async global->LDS copy of one 16-byte chunk (ASYNCcnt-tracked).
// Builtin signature (from hipcc diagnostic): param0 = AS(1) v4i* (printed as
// "__device__" in HIP), param1 = AS(3) v4i*, then imm offset + imm cpol.
// AS pointers are produced via integer casts: global generic addresses are
// identity-mapped; low 32 bits of a generic LDS address are the LDS byte
// offset (ISA 10.2 aperture rules).
// ---------------------------------------------------------------------------
#if defined(__HIP_DEVICE_COMPILE__) && __has_builtin(__builtin_amdgcn_global_load_async_to_lds_b128)
typedef __attribute__((address_space(1))) v4i* glob_v4i_ptr;
typedef __attribute__((address_space(3))) v4i* lds_v4i_ptr;
#endif

__device__ __forceinline__ void async_copy_b128(const bf16* g, bf16* l) {
#if defined(__HIP_DEVICE_COMPILE__) && __has_builtin(__builtin_amdgcn_global_load_async_to_lds_b128)
  __builtin_amdgcn_global_load_async_to_lds_b128(
      (glob_v4i_ptr)(uintptr_t)g,
      (lds_v4i_ptr)(uint32_t)(uintptr_t)l,
      0, 0);
#else
  *(v8bf*)l = *(const v8bf*)g;
#endif
}

__device__ __forceinline__ void wait_async_zero() {
#if defined(__HIP_DEVICE_COMPILE__)
#if __has_builtin(__builtin_amdgcn_s_wait_asynccnt)
  __builtin_amdgcn_s_wait_asynccnt(0);
#else
  asm volatile("s_wait_asynccnt 0" ::: "memory");
#endif
#endif
}

// ---------------------------------------------------------------------------
// fp32 -> bf16 conversion (weights once per launch)
// ---------------------------------------------------------------------------
__global__ __launch_bounds__(256) void k_f32_to_bf16(const float* __restrict__ src,
                                                     bf16* __restrict__ dst, int n) {
  int i = blockIdx.x * 256 + threadIdx.x;
  if (i < n) dst[i] = (bf16)src[i];
}

// strided slice + convert: dst[r, 0:wlen] = src[r, off : off+wlen]
__global__ __launch_bounds__(256) void k_slice_bf16(const float* __restrict__ src,
                                                    bf16* __restrict__ dst,
                                                    int src_row, int off, int wlen, int nrows) {
  int i = blockIdx.x * 256 + threadIdx.x;
  if (i >= nrows * wlen) return;
  int r = i / wlen, c = i - r * wlen;
  dst[i] = (bf16)src[(size_t)r * src_row + off + c];
}

// ---------------------------------------------------------------------------
// Embedding: h[r,d] = sum_{j<4} x[r,j]*emb_w[d,j] + emb_b[d]   (K=4, VALU)
// ---------------------------------------------------------------------------
__global__ __launch_bounds__(256) void k_embed(const float* __restrict__ x,
                                               const float* __restrict__ w,
                                               const float* __restrict__ b,
                                               float* __restrict__ h) {
  int i = blockIdx.x * 256 + threadIdx.x;           // over MROWS*DMODEL
  if (i >= MROWS * DMODEL) return;
  int d = i & (DMODEL - 1);
  int r = i >> 9;
  const float* xr = x + (size_t)r * DIN;
  const float* wr = w + (size_t)d * DIN;
  h[i] = xr[0]*wr[0] + xr[1]*wr[1] + xr[2]*wr[2] + xr[3]*wr[3] + b[d];
}

// ---------------------------------------------------------------------------
// residual += a   /   residual = a
// ---------------------------------------------------------------------------
__global__ __launch_bounds__(256) void k_add(float* __restrict__ dst,
                                             const float* __restrict__ a, int n, int accumulate) {
  int i = blockIdx.x * 256 + threadIdx.x;
  if (i < n) dst[i] = accumulate ? (dst[i] + a[i]) : a[i];
}

// ---------------------------------------------------------------------------
// LayerNorm over DMODEL=512, one block (256 thr) per row, bf16 output
// ---------------------------------------------------------------------------
__global__ __launch_bounds__(256) void k_ln_bf16(const float* __restrict__ in,
                                                 const float* __restrict__ g,
                                                 const float* __restrict__ b,
                                                 bf16* __restrict__ out) {
  __shared__ float s1[256], s2[256];
  int row = blockIdx.x, tid = threadIdx.x;
  const float* xr = in + (size_t)row * DMODEL;
  float v0 = xr[tid], v1 = xr[tid + 256];
  s1[tid] = v0 + v1;
  s2[tid] = v0*v0 + v1*v1;
  __syncthreads();
  for (int o = 128; o; o >>= 1) {
    if (tid < o) { s1[tid] += s1[tid+o]; s2[tid] += s2[tid+o]; }
    __syncthreads();
  }
  float mean = s1[0] * (1.0f / DMODEL);
  float var  = s2[0] * (1.0f / DMODEL) - mean * mean;
  float rstd = rsqrtf(var + 1e-5f);
  bf16* orow = out + (size_t)row * DMODEL;
  orow[tid]       = (bf16)((v0 - mean) * rstd * g[tid]       + b[tid]);
  orow[tid + 256] = (bf16)((v1 - mean) * rstd * g[tid + 256] + b[tid + 256]);
}

// ---------------------------------------------------------------------------
// WMMA GEMM:  Y[M,N] = X[M,K] @ W[N,K]^T   (bf16 in, f32 out)
//
// Block tile 256(M) x 32(N), 8 waves as 4(m) x 2(n); each wave owns a
// 64x16 strip (4 accumulators). Per 32-wide k-step the block stages
// A(256x32) + B(32x32) panels into LDS with ASYNCcnt-tracked
// global_load_async_to_lds_b128, double-buffered: the next panel's async
// copies run while WMMA consumes the current panel out of LDS (ds_load_b128),
// removing the per-wmma s_wait_loadcnt serialization of the L2-direct scheme.
// EPI==1: Y = softplus(Y + bias[n])   (dt projection)
// ---------------------------------------------------------------------------
template<int EPI>
__global__ __launch_bounds__(256) void k_gemm(const bf16* __restrict__ X,
                                              const bf16* __restrict__ W,
                                              const float* __restrict__ bias,
                                              float* __restrict__ Y,
                                              int Mrows, int N, int K) {
  __shared__ __align__(16) bf16 sA[2][256 * 32];   // 2 x 16 KB
  __shared__ __align__(16) bf16 sB[2][32 * 32];    // 2 x  2 KB

  int tiles_n = N >> 5;
  int bm = blockIdx.x / tiles_n;        // 256-row block
  int bn = blockIdx.x - bm * tiles_n;   // 32-col block
  int tid  = threadIdx.x;
  int lane = tid & 31;
  int wave = tid >> 5;
  int wm = wave >> 1;                   // 0..3 : 64-row strip
  int wn = wave & 1;                    // 0..1 : 16-col strip
  int lm = lane & 15, lh = lane >> 4;

  const bf16* Xblk = X + (size_t)(bm * 256) * K;
  const bf16* Wblk = W + (size_t)(bn * 32) * K;

  // stage one 32-wide k-panel into LDS buffer `buf`
  auto issue = [&](int buf, int k0) {
    // A panel: 1024 16-byte chunks, 4 per thread (chunk c covers
    // row c>>2, k-offset (c&3)*8; LDS layout row*32+ko == c*8)
#pragma unroll
    for (int j = 0; j < 4; ++j) {
      int c = tid + j * 256;
      async_copy_b128(Xblk + (size_t)(c >> 2) * K + k0 + (c & 3) * 8,
                      &sA[buf][c * 8]);
    }
    // B panel: 128 16-byte chunks
    if (tid < 128) {
      async_copy_b128(Wblk + (size_t)(tid >> 2) * K + k0 + (tid & 3) * 8,
                      &sB[buf][tid * 8]);
    }
  };

  v8f acc[4] = {};
  int nk = K >> 5;
  issue(0, 0);
  int cur = 0;
  for (int ks = 0; ks < nk; ++ks) {
    wait_async_zero();          // this wave's pending panel copies done
    __syncthreads();            // whole block: panel visible, prior reads done
    if (ks + 1 < nk) issue(cur ^ 1, (ks + 1) << 5);

    // fragments out of LDS (contiguous k = 16*lh + s in both A and B), 4 wmma per k-step
    const bf16* Bb = &sB[cur][(wn * 16 + lm) * 32 + lh * 16];
    v16bf bfrag = *(const v16bf*)Bb;
    const bf16* Ab = &sA[cur][(wm * 64 + lm) * 32 + lh * 16];
#pragma unroll
    for (int rt = 0; rt < 4; ++rt) {
      const bf16* ap = Ab + rt * 16 * 32;
      v8bf alo = *(const v8bf*)(ap);
      v8bf ahi = *(const v8bf*)(ap + 8);
      v16bf afrag = __builtin_shufflevector(alo, ahi,
          0,1,2,3,4,5,6,7,8,9,10,11,12,13,14,15);
      acc[rt] = __builtin_amdgcn_wmma_f32_16x16x32_bf16(
          false, afrag, false, bfrag, (short)0, acc[rt], false, false);
    }
    cur ^= 1;
  }

  int col = bn * 32 + wn * 16 + lm;
  float bv = (EPI == 1) ? bias[col] : 0.0f;
#pragma unroll
  for (int rt = 0; rt < 4; ++rt) {
#pragma unroll
    for (int v = 0; v < 8; ++v) {
      int row = bm * 256 + wm * 64 + rt * 16 + v + lh * 8;  // C/D layout
      float val = acc[rt][v];
      if (EPI == 1) {                                       // softplus(x + b)
        val += bv;
        val = (val > 20.0f) ? val : log1pf(__expf(val));
      }
      Y[(size_t)row * N + col] = val;
    }
  }
}

// ---------------------------------------------------------------------------
// Causal depthwise conv (k=4) over xz[:, :, 0:DINNER] + SiLU; dual f32/bf16 out
// ---------------------------------------------------------------------------
__global__ __launch_bounds__(256) void k_conv_silu(const float* __restrict__ xz,
                                                   const float* __restrict__ cw,
                                                   const float* __restrict__ cb,
                                                   float* __restrict__ xi,
                                                   bf16* __restrict__ xibf) {
  int i = blockIdx.x * 256 + threadIdx.x;          // over MROWS*DINNER
  if (i >= MROWS * DINNER) return;
  int d  = i & (DINNER - 1);
  int bl = i >> 10;
  int l  = bl & (SEQ - 1);
  int b  = bl >> 9;
  const float* w   = cw + (size_t)d * DCONV;
  const float* col = xz + (size_t)b * SEQ * (2 * DINNER) + d;   // row stride 2048
  float s = cb[d];
#pragma unroll
  for (int j = 0; j < DCONV; ++j) {
    int t = l - (DCONV - 1) + j;
    if (t >= 0) s += col[(size_t)t * (2 * DINNER)] * w[j];
  }
  float r = s * (1.0f / (1.0f + __expf(-s)));      // silu
  xi[i] = r;
  xibf[i] = (bf16)r;
}

// ---------------------------------------------------------------------------
// Selective scan: one wave per (b,d) channel, 2 states/lane over DST=64,
// shfl_xor tree reduction per timestep. y += x*Dp fused.
// ---------------------------------------------------------------------------
__global__ __launch_bounds__(256) void k_scan(const float* __restrict__ xi,
                                              const float* __restrict__ dt,
                                              const float* __restrict__ xdbl,
                                              const float* __restrict__ A_log,
                                              const float* __restrict__ Dp,
                                              float* __restrict__ y) {
  int wave = blockIdx.x * 8 + (threadIdx.x >> 5);
  if (wave >= BATCH * DINNER) return;
  int lane = threadIdx.x & 31;
  int b = wave >> 10;
  int d = wave & (DINNER - 1);
  int n0 = lane * 2;
  float A0 = -__expf(A_log[(size_t)d * DSTATE + n0]);
  float A1 = -__expf(A_log[(size_t)d * DSTATE + n0 + 1]);
  float h0 = 0.0f, h1 = 0.0f;
  const float* dtp = dt   + (size_t)b * SEQ * DINNER + d;
  const float* xp  = xi   + (size_t)b * SEQ * DINNER + d;
  const float* bm  = xdbl + (size_t)b * SEQ * 160 + DTRANK + n0;           // B at col 32
  const float* cm  = xdbl + (size_t)b * SEQ * 160 + DTRANK + DSTATE + n0;  // C at col 96
  float Dv = Dp[d];
  float* yp = y + (size_t)b * SEQ * DINNER + d;
  for (int t = 0; t < SEQ; ++t) {
    float dtt = dtp[(size_t)t * DINNER];
    float xt  = xp [(size_t)t * DINNER];
    float2 bv = *(const float2*)(bm + (size_t)t * 160);
    float2 cv = *(const float2*)(cm + (size_t)t * 160);
    float dx = dtt * xt;
    h0 = __expf(dtt * A0) * h0 + dx * bv.x;
    h1 = __expf(dtt * A1) * h1 + dx * bv.y;
    float p = h0 * cv.x + h1 * cv.y;
    for (int o = 16; o; o >>= 1) p += __shfl_xor(p, o, 32);
    if (lane == 0) yp[(size_t)t * DINNER] = p + xt * Dv;
  }
}

// ---------------------------------------------------------------------------
// Gate: out_bf16 = y * silu(z)   (z = xz[:, DINNER:2*DINNER])
// ---------------------------------------------------------------------------
__global__ __launch_bounds__(256) void k_gate(const float* __restrict__ y,
                                              const float* __restrict__ xz,
                                              bf16* __restrict__ out) {
  int i = blockIdx.x * 256 + threadIdx.x;          // over MROWS*DINNER
  if (i >= MROWS * DINNER) return;
  int d = i & (DINNER - 1);
  int r = i >> 10;
  float z = xz[(size_t)r * (2 * DINNER) + DINNER + d];
  float s = z * (1.0f / (1.0f + __expf(-z)));
  out[i] = (bf16)(y[i] * s);
}

// ---------------------------------------------------------------------------
// GLU activation: act = t[:, :HID] * silu(t[:, HID:2HID]) -> bf16
// ---------------------------------------------------------------------------
__global__ __launch_bounds__(256) void k_mlpact(const float* __restrict__ t,
                                                bf16* __restrict__ out) {
  int i = blockIdx.x * 256 + threadIdx.x;          // over MROWS*HIDDEN
  if (i >= MROWS * HIDDEN) return;
  int r = i / HIDDEN;
  int c = i - r * HIDDEN;
  const float* row = t + (size_t)r * (2 * HIDDEN);
  float a = row[c];
  float g = row[HIDDEN + c];
  out[i] = (bf16)(a * g * (1.0f / (1.0f + __expf(-g))));
}

// ---------------------------------------------------------------------------
// Final: LN(h[b, L-1, :]) . head_w + head_b  -> out[b]
// ---------------------------------------------------------------------------
__global__ __launch_bounds__(256) void k_final(const float* __restrict__ h,
                                               const float* __restrict__ g,
                                               const float* __restrict__ be,
                                               const float* __restrict__ hw,
                                               const float* __restrict__ hb,
                                               float* __restrict__ out) {
  __shared__ float s1[256], s2[256];
  int b = blockIdx.x, tid = threadIdx.x;
  const float* xr = h + ((size_t)b * SEQ + (SEQ - 1)) * DMODEL;
  float v0 = xr[tid], v1 = xr[tid + 256];
  s1[tid] = v0 + v1;
  s2[tid] = v0*v0 + v1*v1;
  __syncthreads();
  for (int o = 128; o; o >>= 1) {
    if (tid < o) { s1[tid] += s1[tid+o]; s2[tid] += s2[tid+o]; }
    __syncthreads();
  }
  float mean = s1[0] * (1.0f / DMODEL);
  float var  = s2[0] * (1.0f / DMODEL) - mean * mean;
  float rstd = rsqrtf(var + 1e-5f);
  __syncthreads();
  float n0 = (v0 - mean) * rstd * g[tid]       + be[tid];
  float n1 = (v1 - mean) * rstd * g[tid + 256] + be[tid + 256];
  s1[tid] = n0 * hw[tid] + n1 * hw[tid + 256];
  __syncthreads();
  for (int o = 128; o; o >>= 1) {
    if (tid < o) s1[tid] += s1[tid+o];
    __syncthreads();
  }
  if (tid == 0) out[b] = s1[0] + hb[0];
}

// ---------------------------------------------------------------------------
// host orchestration
// ---------------------------------------------------------------------------
extern "C" void kernel_launch(void* const* d_in, const int* in_sizes, int n_in,
                              void* d_out, int out_size, void* d_ws, size_t ws_size,
                              hipStream_t stream) {
  (void)in_sizes; (void)n_in; (void)out_size; (void)ws_size;
  const float* x       = (const float*)d_in[0];
  const float* emb_w   = (const float*)d_in[1];
  const float* emb_b   = (const float*)d_in[2];
  const float* in_w    = (const float*)d_in[3];
  const float* conv_w  = (const float*)d_in[4];
  const float* conv_b  = (const float*)d_in[5];
  const float* xproj_w = (const float*)d_in[6];
  const float* dt_w    = (const float*)d_in[7];
  const float* dt_b    = (const float*)d_in[8];
  const float* A_log   = (const float*)d_in[9];
  const float* Dp      = (const float*)d_in[10];
  const float* out_w   = (const float*)d_in[11];
  const float* ln1_g   = (const float*)d_in[12];
  const float* ln1_b   = (const float*)d_in[13];
  const float* ln2_g   = (const float*)d_in[14];
  const float* ln2_b   = (const float*)d_in[15];
  const float* fc1_w   = (const float*)d_in[16];
  const float* fc2_w   = (const float*)d_in[17];
  const float* lnf_g   = (const float*)d_in[18];
  const float* lnf_b   = (const float*)d_in[19];
  const float* head_w  = (const float*)d_in[20];
  const float* head_b  = (const float*)d_in[21];
  float* out = (float*)d_out;

  size_t off = 0;
  auto carve = [&](size_t bytes) -> char* {
    char* p = (char*)d_ws + off;
    off += (bytes + 255) & ~(size_t)255;
    return p;
  };
  const size_t M = MROWS;
  float* h        = (float*)carve(M * DMODEL * 4);
  float* residual = (float*)carve(M * DMODEL * 4);
  bf16*  hn_bf    = (bf16*) carve(M * DMODEL * 2);
  float* xz       = (float*)carve(M * 2 * DINNER * 4);
  float* xi       = (float*)carve(M * DINNER * 4);
  bf16*  xi_bf    = (bf16*) carve(M * DINNER * 2);
  float* xdbl     = (float*)carve(M * 160 * 4);
  bf16*  dtr_bf   = (bf16*) carve(M * DTRANK * 2);
  float* dt       = (float*)carve(M * DINNER * 4);
  float* yscan    = (float*)carve(M * DINNER * 4);
  bf16*  yg_bf    = (bf16*) carve(M * DINNER * 2);
  float* hm       = (float*)carve(M * DMODEL * 4);
  float* t_mlp    = (float*)carve(M * 2 * HIDDEN * 4);
  bf16*  act_bf   = (bf16*) carve(M * HIDDEN * 2);
  // bf16 weight mirrors
  const size_t n_in_w  = (size_t)NLAYER * 2 * DINNER * DMODEL;
  const size_t n_xp_w  = (size_t)NLAYER * 160 * DINNER;
  const size_t n_dt_w  = (size_t)NLAYER * DINNER * DTRANK;
  const size_t n_out_w = (size_t)NLAYER * DMODEL * DINNER;
  const size_t n_f1_w  = (size_t)NLAYER * 2 * HIDDEN * DMODEL;
  const size_t n_f2_w  = (size_t)NLAYER * DMODEL * HIDDEN;
  bf16* w_in  = (bf16*)carve(n_in_w  * 2);
  bf16* w_xp  = (bf16*)carve(n_xp_w  * 2);
  bf16* w_dt  = (bf16*)carve(n_dt_w  * 2);
  bf16* w_out = (bf16*)carve(n_out_w * 2);
  bf16* w_f1  = (bf16*)carve(n_f1_w  * 2);
  bf16* w_f2  = (bf16*)carve(n_f2_w  * 2);

  // weight precision conversion (once per launch, deterministic)
  k_f32_to_bf16<<<CDIV((int)n_in_w , 256), 256, 0, stream>>>(in_w,    w_in,  (int)n_in_w);
  k_f32_to_bf16<<<CDIV((int)n_xp_w , 256), 256, 0, stream>>>(xproj_w, w_xp,  (int)n_xp_w);
  k_f32_to_bf16<<<CDIV((int)n_dt_w , 256), 256, 0, stream>>>(dt_w,    w_dt,  (int)n_dt_w);
  k_f32_to_bf16<<<CDIV((int)n_out_w, 256), 256, 0, stream>>>(out_w,   w_out, (int)n_out_w);
  k_f32_to_bf16<<<CDIV((int)n_f1_w , 256), 256, 0, stream>>>(fc1_w,   w_f1,  (int)n_f1_w);
  k_f32_to_bf16<<<CDIV((int)n_f2_w , 256), 256, 0, stream>>>(fc2_w,   w_f2,  (int)n_f2_w);

  auto gemm = [&](const bf16* X, const bf16* Wt, const float* bias, float* Y,
                  int Mr, int N, int K, int epi) {
    int blocks = (Mr >> 8) * (N >> 5);   // 256x32 block tiles
    if (epi == 0) k_gemm<0><<<blocks, 256, 0, stream>>>(X, Wt, bias, Y, Mr, N, K);
    else          k_gemm<1><<<blocks, 256, 0, stream>>>(X, Wt, bias, Y, Mr, N, K);
  };

  const int nMD = MROWS * DMODEL;
  k_embed<<<CDIV(nMD, 256), 256, 0, stream>>>(x, emb_w, emb_b, h);

  for (int i = 0; i < NLAYER; ++i) {
    // residual = h (i==0) else residual += h
    k_add<<<CDIV(nMD, 256), 256, 0, stream>>>(residual, h, nMD, i != 0);
    // hn = LN1(residual) -> bf16
    k_ln_bf16<<<MROWS, 256, 0, stream>>>(residual, ln1_g + i*DMODEL, ln1_b + i*DMODEL, hn_bf);
    // xz = hn @ in_w[i].T        (2048x512 @ 512x2048)
    gemm(hn_bf, w_in + (size_t)i * 2 * DINNER * DMODEL, nullptr, xz, MROWS, 2*DINNER, DMODEL, 0);
    // xi = silu(causal_conv(xz[:, :DINNER]))
    k_conv_silu<<<CDIV(MROWS*DINNER, 256), 256, 0, stream>>>(
        xz, conv_w + (size_t)i*DINNER*DCONV, conv_b + (size_t)i*DINNER, xi, xi_bf);
    // xdbl = xi @ xproj_w[i].T   (2048x1024 @ 1024x160)
    gemm(xi_bf, w_xp + (size_t)i * 160 * DINNER, nullptr, xdbl, MROWS, 160, DINNER, 0);
    // dtr slice -> bf16
    k_slice_bf16<<<CDIV(MROWS*DTRANK, 256), 256, 0, stream>>>(xdbl, dtr_bf, 160, 0, DTRANK, MROWS);
    // dt = softplus(dtr @ dt_w[i].T + dt_b[i])   (2048x32 @ 32x1024)
    gemm(dtr_bf, w_dt + (size_t)i * DINNER * DTRANK, dt_b + (size_t)i*DINNER, dt, MROWS, DINNER, DTRANK, 1);
    // selective scan
    k_scan<<<CDIV(BATCH*DINNER, 8), 256, 0, stream>>>(
        xi, dt, xdbl, A_log + (size_t)i*DINNER*DSTATE, Dp + (size_t)i*DINNER, yscan);
    // y * silu(z) -> bf16
    k_gate<<<CDIV(MROWS*DINNER, 256), 256, 0, stream>>>(yscan, xz, yg_bf);
    // hm = yg @ out_w[i].T       (2048x1024 @ 1024x512)
    gemm(yg_bf, w_out + (size_t)i * DMODEL * DINNER, nullptr, hm, MROWS, DMODEL, DINNER, 0);
    // residual += hm
    k_add<<<CDIV(nMD, 256), 256, 0, stream>>>(residual, hm, nMD, 1);
    // hn2 = LN2(residual) -> bf16
    k_ln_bf16<<<MROWS, 256, 0, stream>>>(residual, ln2_g + i*DMODEL, ln2_b + i*DMODEL, hn_bf);
    // t = hn2 @ fc1_w[i].T       (2048x512 @ 512x2816)
    gemm(hn_bf, w_f1 + (size_t)i * 2 * HIDDEN * DMODEL, nullptr, t_mlp, MROWS, 2*HIDDEN, DMODEL, 0);
    // act = t[:, :HID] * silu(t[:, HID:]) -> bf16
    k_mlpact<<<CDIV(MROWS*HIDDEN, 256), 256, 0, stream>>>(t_mlp, act_bf);
    // h = act @ fc2_w[i].T       (2048x1408 @ 1408x512)
    gemm(act_bf, w_f2 + (size_t)i * DMODEL * HIDDEN, nullptr, h, MROWS, DMODEL, HIDDEN, 0);
  }

  // final LN + head on last token
  k_final<<<BATCH, 256, 0, stream>>>(h, lnf_g, lnf_b, head_w, head_b, out);
}
